// ChebNet_64991445123374
// MI455X (gfx1250) — matrix operand order repaired and checked by reading, not compiled, not run on hardware
//
#include <hip/hip_runtime.h>
#include <hip/hip_bf16.h>
#include <stdint.h>

// ---------------------------------------------------------------------------
// ChebNet forward for MI455X (gfx1250, wave32).
//  - Sparse props: L2-resident gather + HW f32 scatter atomics (edge-coalesced).
//  - Dense transform: single [N x 5*Fin] @ [5*Fin x 48] GEMM per layer using
//    v_wmma_f32_16x16x32_bf16; A/B kept as pre-packed bf16 so the hot loop is
//    pure loads + WMMA; weight tile staged to LDS via the Tensor Data Mover.
// ---------------------------------------------------------------------------

typedef __attribute__((ext_vector_type(16))) __bf16        v16bf;
typedef __attribute__((ext_vector_type(8)))  float         v8f;
typedef __attribute__((ext_vector_type(4)))  unsigned int  v4u;
typedef __attribute__((ext_vector_type(8)))  int           v8i;
typedef __attribute__((ext_vector_type(4)))  int           v4i;

union Frag { v16bf v; unsigned u[8]; };

__device__ __forceinline__ unsigned short f2bf(float f) {
  union { float f; unsigned u; } x; x.f = f;
  unsigned r = x.u + 0x7FFFu + ((x.u >> 16) & 1u);   // round-to-nearest-even
  return (unsigned short)(r >> 16);
}

__device__ __forceinline__ void atomAddF(float* p, float v) { unsafeAtomicAdd(p, v); }

// ------------------------------- utility kernels ---------------------------

__global__ void fill0_k(float* __restrict__ p, unsigned n) {
  unsigned i = blockIdx.x * blockDim.x + threadIdx.x;
  if (i < n) p[i] = 0.0f;
}

// pack pairs of f32 -> one u32 of two bf16 (coalesced)
__global__ void cvt_pack_k(const float* __restrict__ in,
                           unsigned short* __restrict__ outb, unsigned npairs) {
  unsigned i = blockIdx.x * blockDim.x + threadIdx.x;
  if (i >= npairs) return;
  float2 v = ((const float2*)in)[i];
  ((unsigned*)outb)[i] = (unsigned)f2bf(v.x) | ((unsigned)f2bf(v.y) << 16);
}

__global__ void deg_k(const int* __restrict__ src, float* __restrict__ deg, int E) {
  unsigned e = blockIdx.x * blockDim.x + threadIdx.x;
  if (e < (unsigned)E) atomAddF(&deg[src[e]], 1.0f);
}

__global__ void wedge_k(const int* __restrict__ src, const int* __restrict__ dst,
                        const int* __restrict__ batch, const float* __restrict__ lmax,
                        const float* __restrict__ deg, float* __restrict__ wedge, int E) {
  unsigned e = blockIdx.x * blockDim.x + threadIdx.x;
  if (e >= (unsigned)E) return;
  int s = src[e], d = dst[e];
  float ds = deg[s], dd = deg[d];
  float dis = ds > 0.0f ? __frsqrt_rn(ds) : 0.0f;
  float did = dd > 0.0f ? __frsqrt_rn(dd) : 0.0f;
  wedge[e] = -2.0f * dis * did / lmax[batch[s]];
}

__global__ void diag_k(const int* __restrict__ batch, const float* __restrict__ lmax,
                       float* __restrict__ diag, int N) {
  unsigned n = blockIdx.x * blockDim.x + threadIdx.x;
  if (n < (unsigned)N) diag[n] = 2.0f / lmax[batch[n]] - 1.0f;
}

// Pack all 5 layers' weights: Wb[l][n][kk] (bf16 bits), kk = k*Fin + fi, padded to 256.
__global__ void build_w_k(const float* __restrict__ W1, const float* __restrict__ W2,
                          const float* __restrict__ W3, const float* __restrict__ W4,
                          const float* __restrict__ W5, unsigned short* __restrict__ Wb) {
  unsigned i = blockIdx.x * blockDim.x + threadIdx.x;
  if (i >= 5u * 48u * 256u) return;
  int l = i / (48 * 256);
  int r = i % (48 * 256);
  int n = r >> 8;            // output column 0..47
  int kk = r & 255;          // padded K index
  int FIN = (l == 0) ? 16 : 48;
  int KD = 5 * FIN;
  float val = 0.0f;
  if (kk < KD) {
    int k = kk / FIN, fi = kk % FIN;
    const float* W = (l == 0) ? W1 : (l == 1) ? W2 : (l == 2) ? W3 : (l == 3) ? W4 : W5;
    val = W[((size_t)k * FIN + fi) * 48 + n];
  }
  Wb[i] = f2bf(val);
}

// ------------------------------- sparse prop -------------------------------

// One thread per (edge, 4-feature chunk): coalesced gather of T[src] row,
// coalesced burst of HW f32 atomics into S[dst] row. S must be pre-zeroed.
template <int F>
__global__ void scatter_k(const int* __restrict__ src, const int* __restrict__ dst,
                          const float* __restrict__ wedge, const float* __restrict__ T,
                          float* __restrict__ S, int E) {
  constexpr int C = F / 4;
  unsigned i = blockIdx.x * blockDim.x + threadIdx.x;
  unsigned total = (unsigned)E * C;
  if (i >= total) return;
  int e = i / C;
  int c = (i % C) * 4;
  int s = src[e], d = dst[e];
  float w = wedge[e];
  float4 t = *(const float4*)(T + (size_t)s * F + c);
  float* o = S + (size_t)d * F + c;
  atomAddF(o + 0, w * t.x);
  atomAddF(o + 1, w * t.y);
  atomAddF(o + 2, w * t.z);
  atomAddF(o + 3, w * t.w);
}

// Tout = a*(S + diag[n]*Tprev) + bco*Told ; also emits packed-bf16 shadow copy.
template <int F>
__global__ void combine_k(const float* __restrict__ S, const float* __restrict__ Tprev,
                          const float* __restrict__ Told, const float* __restrict__ diag,
                          float a, float bco, float* __restrict__ Tout,
                          unsigned short* __restrict__ Toutb, int N) {
  unsigned i = blockIdx.x * blockDim.x + threadIdx.x;
  unsigned total = (unsigned)N * (F / 2);
  if (i >= total) return;
  unsigned n = (2u * i) / F;
  float2 s  = ((const float2*)S)[i];
  float2 tp = ((const float2*)Tprev)[i];
  float2 to = ((const float2*)Told)[i];
  float d = diag[n];
  float rx = a * (s.x + d * tp.x) + bco * to.x;
  float ry = a * (s.y + d * tp.y) + bco * to.y;
  ((float2*)Tout)[i] = make_float2(rx, ry);
  ((unsigned*)Toutb)[i] = (unsigned)f2bf(rx) | ((unsigned)f2bf(ry) << 16);
}

// ------------------------------- WMMA GEMM ---------------------------------

// out[n, 0..47] = relu( bias + concat(T0..T4)[n,:] @ Wstack )
// Wave-level: 16-row M tile x three 16-col N tiles, K chunks of 32 (bf16).
template <int FIN>
__global__ void __launch_bounds__(256) cheb_gemm_k(
    const unsigned short* __restrict__ Tb0, const unsigned short* __restrict__ Tb1,
    const unsigned short* __restrict__ Tb2, const unsigned short* __restrict__ Tb3,
    const unsigned short* __restrict__ Tb4, const unsigned short* __restrict__ Wb,
    const float* __restrict__ bias, float* __restrict__ out, int nrows) {
  constexpr int KD = 5 * FIN;            // 80 or 240
  constexpr int NCH = (KD + 31) / 32;    // 3 or 8 K-chunks
  constexpr unsigned TILE_ELEMS = 48u * 256u;  // bf16 elements in weight tile

  __shared__ alignas(16) unsigned short Bsh[TILE_ELEMS];  // [N][Kpad] bf16 bits

  const unsigned tid = threadIdx.x;
  const unsigned lane = tid & 31u;
  const unsigned wave = tid >> 5;

#if __has_builtin(__builtin_amdgcn_tensor_load_to_lds) && \
    __has_builtin(__builtin_amdgcn_s_wait_tensorcnt)
  // --- stage weight tile via Tensor Data Mover (one DMA per block) ---
  if (wave == 0) {
    const unsigned lds_off = (unsigned)(uintptr_t)(void*)&Bsh[0];
    const unsigned long long ga = (unsigned long long)(uintptr_t)Wb;
    v4u g0;
    g0[0] = 1u;                                   // count=1 (valid user D#)
    g0[1] = lds_off;                              // lds_addr (bytes)
    g0[2] = (unsigned)ga;                         // global_addr[31:0]
    g0[3] = (unsigned)((ga >> 32) & 0x1FFFFFFull) // global_addr[56:32]
            | 0x80000000u;                        // type=2 ("image")
    v8i g1;
    g1[0] = 0x00010000;                           // data_size=2B; mask=0
    g1[1] = (int)((TILE_ELEMS & 0xFFFFu) << 16);  // tensor_dim0[15:0]
    g1[2] = 0x00010000;                           // tensor_dim0[31:16]=0, tensor_dim1=1
    g1[3] = (int)((TILE_ELEMS & 0xFFFFu) << 16);  // tile_dim0 = whole 1-D tile
    g1[4] = 0;                                    // tile_dim1=0, tile_dim2=0
    g1[5] = (int)TILE_ELEMS;                      // tensor_dim0_stride lo32
    g1[6] = 0;
    g1[7] = 0;
    v4i gz  = {0, 0, 0, 0};                       // groups 2/3 unused (<=2D)
    v8i gz8 = {0, 0, 0, 0, 0, 0, 0, 0};          // extra group (6-arg form)
    __builtin_amdgcn_tensor_load_to_lds(g0, g1, gz, gz, gz8, 0);
    __builtin_amdgcn_s_wait_tensorcnt(0);
  }
#else
  {  // fallback: cooperative coalesced copy
    const uint2* wsrc = (const uint2*)Wb;
    uint2* wdst = (uint2*)Bsh;
    for (unsigned i = tid; i < TILE_ELEMS / 4u; i += 256u) wdst[i] = wsrc[i];
  }
#endif
  __syncthreads();

  const int r0 = blockIdx.x * 128 + (int)wave * 16;
  if (r0 >= nrows) return;  // wave-uniform: EXEC stays all-ones for WMMA

  const int arow = r0 + (int)(lane & 15u);
  const int klo = (lane < 16u) ? 0 : 8;      // A-fragment K sublane offset
  const int kboff = (lane < 16u) ? 0 : 8;    // B-fragment K sublane offset
  const int nlocal = (int)(lane & 15u);

  v8f acc[3] = {{}, {}, {}};

#pragma unroll
  for (int kc = 0; kc < NCH; ++kc) {
    // ---- A fragment: 16x32 bf16 per ISA layout; pre-packed pairs, so each
    //      VGPR is one aligned 32-bit load (v=0..3 / v=4..7 are contiguous) ----
    Frag a;
#pragma unroll
    for (int v = 0; v < 8; ++v) {
      int kk = kc * 32 + ((v < 4) ? 0 : 16) + klo + 2 * (v & 3);  // even
      unsigned pr = 0u;
      if (kk < KD) {  // pair (kk,kk+1) stays inside one T row (FIN even)
        int t = kk / FIN;
        int col = kk - t * FIN;
        const unsigned short* Tt = (t == 0) ? Tb0 : (t == 1) ? Tb1 : (t == 2) ? Tb2
                                  : (t == 3) ? Tb3 : Tb4;
        pr = *(const unsigned*)(Tt + (size_t)arow * FIN + col);
      }
      a.u[v] = pr;
    }
    // ---- B fragments + WMMA for the three 16-wide N tiles ----
#pragma unroll
    for (int nt = 0; nt < 3; ++nt) {
      Frag b;
      const int nb = (nt * 16 + nlocal) * 256 + kc * 32 + kboff;
#pragma unroll
      for (int v = 0; v < 8; ++v) {
        b.u[v] = *(const unsigned*)&Bsh[nb + ((v < 4) ? 0 : 16) + 2 * (v & 3)];  // 4B aligned, 32B/cluster
      }
      acc[nt] = __builtin_amdgcn_wmma_f32_16x16x32_bf16(
          false, a.v, false, b.v, (short)0, acc[nt], false, false);
    }
  }

  // ---- epilogue: bias + relu, f32 stores (C/D layout: VGPR r <-> M=r / 8+r) ----
  const int rbase = r0 + ((lane < 16u) ? 0 : 8);
#pragma unroll
  for (int nt = 0; nt < 3; ++nt) {
    const int col = nt * 16 + nlocal;
    const float bv = bias[col];
#pragma unroll
    for (int r = 0; r < 8; ++r) {
      float vv = acc[nt][r] + bv;
      out[(size_t)(rbase + r) * 48 + col] = fmaxf(vv, 0.0f);
    }
  }
}

// ------------------------------- pooling & head ----------------------------

__global__ void pool_k(const float* __restrict__ h, const int* __restrict__ batch,
                       float* __restrict__ pooled, int N) {
  unsigned i = blockIdx.x * blockDim.x + threadIdx.x;
  unsigned total = (unsigned)N * 48u;
  if (i >= total) return;
  unsigned n = i / 48u;
  unsigned f = i - n * 48u;
  atomAddF(&pooled[(size_t)batch[n] * 48 + f], h[i]);
}

__global__ void head_k(const float* __restrict__ pooled, const float* __restrict__ fc1w,
                       const float* __restrict__ fc1b, const float* __restrict__ fc2w,
                       const float* __restrict__ fc2b, float* __restrict__ out) {
  int g = blockIdx.x;
  int j = threadIdx.x;  // 32 threads = 1 wave
  float s = fc1b[j];
#pragma unroll
  for (int i = 0; i < 48; ++i) s = fmaf(pooled[g * 48 + i], fc1w[i * 32 + j], s);
  s = fmaxf(s, 0.0f) * fc2w[j];
#pragma unroll
  for (int off = 16; off > 0; off >>= 1) s += __shfl_down(s, off);
  if (j == 0) out[g] = s + fc2b[0];
}

// ------------------------------- host side ---------------------------------

static inline unsigned gridFor(long n, int bs = 256) {
  return (unsigned)((n + bs - 1) / bs);
}

template <int F>
static void run_layer(const float* in, unsigned short* inb, const unsigned short* Wb_l,
                      const float* bias, const int* src, const int* dst,
                      const float* wedge, const float* diag, float* S,
                      float* T1, float* T2, float* T3, float* T4,
                      unsigned short* Tb1, unsigned short* Tb2, unsigned short* Tb3,
                      unsigned short* Tb4, float* out, int N, int E,
                      hipStream_t stream) {
  const long nf = (long)N * F;
  const long sc = (long)E * (F / 4);

  cvt_pack_k<<<gridFor(nf / 2), 256, 0, stream>>>(in, inb, (unsigned)(nf / 2));

  fill0_k<<<gridFor(nf), 256, 0, stream>>>(S, (unsigned)nf);
  scatter_k<F><<<gridFor(sc), 256, 0, stream>>>(src, dst, wedge, in, S, E);
  combine_k<F><<<gridFor(nf / 2), 256, 0, stream>>>(S, in, in, diag, 1.0f, 0.0f,
                                                    T1, Tb1, N);

  fill0_k<<<gridFor(nf), 256, 0, stream>>>(S, (unsigned)nf);
  scatter_k<F><<<gridFor(sc), 256, 0, stream>>>(src, dst, wedge, T1, S, E);
  combine_k<F><<<gridFor(nf / 2), 256, 0, stream>>>(S, T1, in, diag, 2.0f, -1.0f,
                                                    T2, Tb2, N);

  fill0_k<<<gridFor(nf), 256, 0, stream>>>(S, (unsigned)nf);
  scatter_k<F><<<gridFor(sc), 256, 0, stream>>>(src, dst, wedge, T2, S, E);
  combine_k<F><<<gridFor(nf / 2), 256, 0, stream>>>(S, T2, T1, diag, 2.0f, -1.0f,
                                                    T3, Tb3, N);

  fill0_k<<<gridFor(nf), 256, 0, stream>>>(S, (unsigned)nf);
  scatter_k<F><<<gridFor(sc), 256, 0, stream>>>(src, dst, wedge, T3, S, E);
  combine_k<F><<<gridFor(nf / 2), 256, 0, stream>>>(S, T3, T2, diag, 2.0f, -1.0f,
                                                    T4, Tb4, N);

  cheb_gemm_k<F><<<(N + 127) / 128, 256, 0, stream>>>(inb, Tb1, Tb2, Tb3, Tb4, Wb_l,
                                                      bias, out, N);
}

extern "C" void kernel_launch(void* const* d_in, const int* in_sizes, int n_in,
                              void* d_out, int out_size, void* d_ws, size_t ws_size,
                              hipStream_t stream) {
  const float* x     = (const float*)d_in[0];
  const int*   eidx  = (const int*)d_in[1];
  const int*   batch = (const int*)d_in[2];
  const float* lmax  = (const float*)d_in[3];
  const float* W[5]  = {(const float*)d_in[4], (const float*)d_in[6],
                        (const float*)d_in[8], (const float*)d_in[10],
                        (const float*)d_in[12]};
  const float* B[5]  = {(const float*)d_in[5], (const float*)d_in[7],
                        (const float*)d_in[9], (const float*)d_in[11],
                        (const float*)d_in[13]};
  const float* fc1w  = (const float*)d_in[14];
  const float* fc1b  = (const float*)d_in[15];
  const float* fc2w  = (const float*)d_in[16];
  const float* fc2b  = (const float*)d_in[17];
  float*       out   = (float*)d_out;

  const int N = in_sizes[0] / 16;  // 100000
  const int E = in_sizes[1] / 2;   // 3200000
  const int G = in_sizes[3];       // 100

  const int* src = eidx;
  const int* dst = eidx + E;

  // ---- workspace carve (256B aligned) ----
  char* p = (char*)d_ws;
  auto carve = [&](size_t bytes) -> char* {
    char* r = p;
    p += (bytes + 255) & ~(size_t)255;
    return r;
  };
  float* deg    = (float*)carve((size_t)N * 4);
  float* diag   = (float*)carve((size_t)N * 4);
  float* wedge  = (float*)carve((size_t)E * 4);
  float* S      = (float*)carve((size_t)N * 48 * 4);
  float* T1     = (float*)carve((size_t)N * 48 * 4);
  float* T2     = (float*)carve((size_t)N * 48 * 4);
  float* T3     = (float*)carve((size_t)N * 48 * 4);
  float* T4     = (float*)carve((size_t)N * 48 * 4);
  float* hA     = (float*)carve((size_t)N * 48 * 4);
  float* hB     = (float*)carve((size_t)N * 48 * 4);
  unsigned short* inb = (unsigned short*)carve((size_t)N * 48 * 2);
  unsigned short* Tb1 = (unsigned short*)carve((size_t)N * 48 * 2);
  unsigned short* Tb2 = (unsigned short*)carve((size_t)N * 48 * 2);
  unsigned short* Tb3 = (unsigned short*)carve((size_t)N * 48 * 2);
  unsigned short* Tb4 = (unsigned short*)carve((size_t)N * 48 * 2);
  unsigned short* Wb  = (unsigned short*)carve((size_t)5 * 48 * 256 * 2);
  float* pooled = (float*)carve((size_t)G * 48 * 4);

  // ---- graph normalization precompute ----
  fill0_k<<<gridFor(N), 256, 0, stream>>>(deg, (unsigned)N);
  deg_k<<<gridFor(E), 256, 0, stream>>>(src, deg, E);
  wedge_k<<<gridFor(E), 256, 0, stream>>>(src, dst, batch, lmax, deg, wedge, E);
  diag_k<<<gridFor(N), 256, 0, stream>>>(batch, lmax, diag, N);
  build_w_k<<<gridFor(5 * 48 * 256), 256, 0, stream>>>(W[0], W[1], W[2], W[3], W[4], Wb);

  // ---- 5 ChebConv layers ----
  run_layer<16>(x, inb, Wb + 0 * 48 * 256, B[0], src, dst, wedge, diag, S,
                T1, T2, T3, T4, Tb1, Tb2, Tb3, Tb4, hA, N, E, stream);
  run_layer<48>(hA, inb, Wb + 1 * 48 * 256, B[1], src, dst, wedge, diag, S,
                T1, T2, T3, T4, Tb1, Tb2, Tb3, Tb4, hB, N, E, stream);
  run_layer<48>(hB, inb, Wb + 2 * 48 * 256, B[2], src, dst, wedge, diag, S,
                T1, T2, T3, T4, Tb1, Tb2, Tb3, Tb4, hA, N, E, stream);
  run_layer<48>(hA, inb, Wb + 3 * 48 * 256, B[3], src, dst, wedge, diag, S,
                T1, T2, T3, T4, Tb1, Tb2, Tb3, Tb4, hB, N, E, stream);
  run_layer<48>(hB, inb, Wb + 4 * 48 * 256, B[4], src, dst, wedge, diag, S,
                T1, T2, T3, T4, Tb1, Tb2, Tb3, Tb4, hA, N, E, stream);

  // ---- pool + MLP head ----
  fill0_k<<<gridFor((long)G * 48), 256, 0, stream>>>(pooled, (unsigned)(G * 48));
  pool_k<<<gridFor((long)N * 48), 256, 0, stream>>>(hA, batch, pooled, N);
  head_k<<<G, 32, 0, stream>>>(pooled, fc1w, fc1b, fc2w, fc2b, out);
}
